// FusedThorExpert_8899172237665
// MI455X (gfx1250) — matrix operand order repaired and checked by reading, not compiled, not run on hardware
//
#include <hip/hip_runtime.h>

typedef _Float16 half_t;
typedef __attribute__((ext_vector_type(16))) _Float16 v16h;
typedef __attribute__((ext_vector_type(8)))  _Float16 v8h;
typedef __attribute__((ext_vector_type(4)))  _Float16 v4h;
typedef __attribute__((ext_vector_type(8)))  float    v8f;

#define E_NUM 16
#define H_DIM 1024
#define I_DIM 4096
#define BM 128
#define BN 128
#define BK 32
#define TPB 256

// prefix-sum of CAPS from the reference (compile-time constants)
__constant__ int c_off[E_NUM + 1] = {
    0, 1024, 1536, 2560, 2816, 3840, 4352, 4480, 5504,
    6016, 7040, 7296, 7808, 8832, 8960, 9024, 9536};

__device__ __forceinline__ void wait_async0() {
#if __has_builtin(__builtin_amdgcn_s_wait_asynccnt)
  __builtin_amdgcn_s_wait_asynccnt(0);
#else
  asm volatile("s_wait_asynccnt 0x0" ::: "memory");
#endif
}

// CDNA5 async copy: global (16B/lane) -> LDS, tracked by ASYNCcnt, no VGPR data.
__device__ __forceinline__ void async_g2l_b128(unsigned lds_addr,
                                               unsigned long long gaddr) {
  asm volatile("global_load_async_to_lds_b128 %0, %1, off"
               :: "v"(lds_addr), "v"(gaddr)
               : "memory");
}

// NT GEMM: Out[m,n] = act( sum_k A[m,k] * W[e,n,k] + bias[e,n] )
// A: tokens x KDIM (AT = float for x, half for h), W: E x Ndim x KDIM (fp32),
// Out: tokens x Ndim (OT = half for h-ws, float for final out).
template <int KDIM, bool FUSE_GELU, typename AT, typename OT>
__global__ __launch_bounds__(TPB) void moe_gemm_kernel(
    const AT* __restrict__ A, const float* __restrict__ W,
    const float* __restrict__ bias, OT* __restrict__ Out, int Ndim) {
  constexpr bool ASYNC_A = (sizeof(AT) == 2);   // f16 A: DMA straight to LDS

  const int e   = blockIdx.z;
  const int off = c_off[e];
  const int cap = c_off[e + 1] - off;
  const int m0  = blockIdx.y * BM;
  if (m0 >= cap) return;               // block-uniform exit (EXEC stays full)
  const int n0  = blockIdx.x * BN;

  const int tid   = threadIdx.x;
  const int lane  = tid & 31;
  const int wave  = tid >> 5;
  const int wm    = wave >> 2;         // 0..1  (M direction)
  const int wn    = wave & 3;          // 0..3  (N direction)
  const int lhalf = lane >> 4;         // 0/1
  const int l16   = lane & 15;

  __shared__ half_t sA[2][BM][BK];     // 16 KB
  __shared__ half_t sB[2][BN][BK];     // 16 KB

  v8f acc[4][2] = {};                  // 4 M-frags x 2 N-frags per wave

  const float* Wb = W + (size_t)e * Ndim * KDIM + (size_t)n0 * KDIM;
  const AT*    Ab = A + (size_t)off * KDIM;

  constexpr int KT = KDIM / BK;

  float4 regB[4];
  float4 regAf[4];

  // Issues loads for stage kt. Async path writes LDS[dstbuf] directly.
  auto load_stage = [&](int kt, int dstbuf) {
    const int k0 = kt * BK;
#pragma unroll
    for (int i = 0; i < 4; ++i) {      // W tile: 128 rows x 32 f32
      int chunk = tid + i * TPB;
      int row = chunk >> 3;
      int kc  = chunk & 7;
      regB[i] = *(const float4*)(Wb + (size_t)row * KDIM + k0 + kc * 4);
    }
    if constexpr (!ASYNC_A) {          // x tile: fp32 via registers (needs cvt)
#pragma unroll
      for (int i = 0; i < 4; ++i) {
        int chunk = tid + i * TPB;
        int row = chunk >> 3;
        int kc  = chunk & 7;
        int gr  = m0 + row; if (gr >= cap) gr = cap - 1;   // clamp OOB rows
        regAf[i] = *(const float4*)((const float*)Ab + (size_t)gr * KDIM + k0 + kc * 4);
      }
    } else {                           // h tile: f16, async DMA global->LDS
#pragma unroll
      for (int i = 0; i < 2; ++i) {
        int chunk = tid + i * TPB;     // 0..511 chunks of 8 halves (16B)
        int row = chunk >> 2;
        int kc  = chunk & 3;
        int gr  = m0 + row; if (gr >= cap) gr = cap - 1;
        unsigned long long g =
            (unsigned long long)(const half_t*)(Ab) +
            2ull * ((size_t)gr * KDIM + k0 + kc * 8);
        unsigned l = (unsigned)(size_t)&sA[dstbuf][row][kc * 8];
        async_g2l_b128(l, g);
      }
    }
  };

  auto store_stage = [&](int buf) {
#pragma unroll
    for (int i = 0; i < 4; ++i) {
      int chunk = tid + i * TPB;
      int row = chunk >> 3;
      int kc  = chunk & 7;
      v4h hv = {(half_t)regB[i].x, (half_t)regB[i].y,
                (half_t)regB[i].z, (half_t)regB[i].w};
      *(v4h*)&sB[buf][row][kc * 4] = hv;
    }
    if constexpr (!ASYNC_A) {
#pragma unroll
      for (int i = 0; i < 4; ++i) {
        int chunk = tid + i * TPB;
        int row = chunk >> 3;
        int kc  = chunk & 7;
        v4h hv = {(half_t)regAf[i].x, (half_t)regAf[i].y,
                  (half_t)regAf[i].z, (half_t)regAf[i].w};
        *(v4h*)&sA[buf][row][kc * 4] = hv;
      }
    }
  };

  auto compute_stage = [&](int buf) {
    v16h afr[4], bfr[2];
    // A 16x32 f16 frag: lane(l16,lhalf) holds K = {8*lhalf..+8} U {16+8*lhalf..+8}
#pragma unroll
    for (int i = 0; i < 4; ++i) {
      const half_t* p = &sA[buf][wm * 64 + i * 16 + l16][lhalf * 8];
      v8h lo = *(const v8h*)(p);
      v8h hi = *(const v8h*)(p + 16);
      afr[i] = __builtin_shufflevector(lo, hi, 0, 1, 2, 3, 4, 5, 6, 7,
                                       8, 9, 10, 11, 12, 13, 14, 15);
    }
    // B 32x16 f16 frag: lane(l16=n, lhalf) holds K = {8*lhalf..+8} U {16+8*lhalf..+8}
#pragma unroll
    for (int j = 0; j < 2; ++j) {
      const half_t* p = &sB[buf][wn * 32 + j * 16 + l16][lhalf * 8];
      v8h lo = *(const v8h*)(p);
      v8h hi = *(const v8h*)(p + 16);
      bfr[j] = __builtin_shufflevector(lo, hi, 0, 1, 2, 3, 4, 5, 6, 7,
                                       8, 9, 10, 11, 12, 13, 14, 15);
    }
#pragma unroll
    for (int i = 0; i < 4; ++i)
#pragma unroll
      for (int j = 0; j < 2; ++j)
        acc[i][j] = __builtin_amdgcn_wmma_f32_16x16x32_f16(
            false, afr[i], false, bfr[j], (short)0, acc[i][j], false, false);
  };

  // Prologue: stage 0 into buffer 0
  load_stage(0, 0);
  store_stage(0);
  if constexpr (ASYNC_A) wait_async0();
  __syncthreads();

#pragma unroll 1
  for (int kt = 0; kt < KT; ++kt) {
    const int buf = kt & 1;
    if (kt + 2 < KT) {                 // L2 prefetch of stage kt+2 weights
      int prow = tid >> 1;
      int pk   = (tid & 1) * 16;
      __builtin_prefetch(Wb + (size_t)prow * KDIM + (kt + 2) * BK + pk, 0, 1);
    }
    if (kt + 1 < KT) load_stage(kt + 1, buf ^ 1);  // overlap with WMMA below
    compute_stage(buf);
    if (kt + 1 < KT) {
      store_stage(buf ^ 1);
      if constexpr (ASYNC_A) wait_async0();        // LDS writes visible...
      __syncthreads();                             // ...to all waves
    }
  }

  // Epilogue: D[m = v + 8*lhalf][n = l16] per frag (ISA C/D layout)
#pragma unroll
  for (int i = 0; i < 4; ++i) {
#pragma unroll
    for (int j = 0; j < 2; ++j) {
      const int n_g  = n0 + wn * 32 + j * 16 + l16;
      const float bv = bias[(size_t)e * Ndim + n_g];
#pragma unroll
      for (int v = 0; v < 8; ++v) {
        const int mg = m0 + wm * 64 + i * 16 + 8 * lhalf + v;
        if (mg < cap) {
          float r = acc[i][j][v] + bv;
          if constexpr (FUSE_GELU)
            r = 0.5f * r * (1.0f + erff(r * 0.70710678118654752f));
          Out[(size_t)(off + mg) * Ndim + n_g] = (OT)r;
        }
      }
    }
  }
}

extern "C" void kernel_launch(void* const* d_in, const int* in_sizes, int n_in,
                              void* d_out, int out_size, void* d_ws, size_t ws_size,
                              hipStream_t stream) {
  const float* x  = (const float*)d_in[0];   // 9536 x 1024
  const float* W1 = (const float*)d_in[1];   // 16 x 4096 x 1024
  const float* b1 = (const float*)d_in[2];   // 16 x 4096
  const float* W2 = (const float*)d_in[3];   // 16 x 1024 x 4096
  const float* b2 = (const float*)d_in[4];   // 16 x 1024
  float*  out  = (float*)d_out;              // 9536 x 1024
  half_t* h_ws = (half_t*)d_ws;              // 9536 x 4096 f16 = 78 MB

  dim3 blk(TPB, 1, 1);
  // Phase 1: h = gelu(x @ W1^T + b1), f16 into workspace
  dim3 g1(I_DIM / BN, 8, E_NUM);
  moe_gemm_kernel<H_DIM, true, float, half_t>
      <<<g1, blk, 0, stream>>>(x, W1, b1, h_ws, I_DIM);
  // Phase 2: out = h @ W2^T + b2, fp32 (A streamed via async-to-LDS DMA)
  dim3 g2(H_DIM / BN, 8, E_NUM);
  moe_gemm_kernel<I_DIM, false, half_t, float>
      <<<g2, blk, 0, stream>>>(h_ws, W2, b2, out, H_DIM);
}
